// SlidingWindowAttention_24764781429074
// MI455X (gfx1250) — matrix operand order repaired and checked by reading, not compiled, not run on hardware
//
#include <hip/hip_runtime.h>
#include <hip/hip_bf16.h>

typedef __bf16 bf16;
typedef __attribute__((ext_vector_type(16))) __bf16 v16bf;
typedef __attribute__((ext_vector_type(8)))  __bf16 v8bf;
typedef __attribute__((ext_vector_type(8)))  float  v8f;

#define T_SEQ  2048
#define DMODEL 1024
#define NH     16
#define HD     64
#define BATCH  2
#define MROWS  (BATCH * T_SEQ)          // 4096
#define N_QKV  (3 * DMODEL)             // 3072
#define PLDS_STRIDE 40                  // 16x32 P tile, padded rows (80B, 16B-mult)

__device__ __forceinline__ v16bf pack2(v8bf lo, v8bf hi) {
  v16bf r;
#pragma unroll
  for (int i = 0; i < 8; ++i) { r[i] = lo[i]; r[i + 8] = hi[i]; }
  return r;
}

__device__ __forceinline__ v8f wmma_bf16(v16bf a, v16bf b, v8f c) {
  return __builtin_amdgcn_wmma_f32_16x16x32_bf16(false, a, false, b, (short)0, c,
                                                 false, false);
}

__device__ __forceinline__ v8f vzero8() {
  v8f z;
#pragma unroll
  for (int i = 0; i < 8; ++i) z[i] = 0.0f;
  return z;
}

// ---------------------------------------------------------------- converts
__global__ void cvt_f32_bf16(const float* __restrict__ src,
                             bf16* __restrict__ dst, int n) {
  int i = blockIdx.x * blockDim.x + threadIdx.x;
  int stride = gridDim.x * blockDim.x;
  for (; i < n; i += stride) dst[i] = (bf16)src[i];
}

// ---------------------------------------------------------------- QKV GEMM
// qkv[m,n] = sum_k S[m,k] * Wqkv[n,k] + b[n].
// 2x4 register blocking (32x64 per wave). waves_per_eu(4,8) relaxes the
// occupancy-driven VGPR cap (128) so the scheduler can double-buffer
// fragments across K-steps instead of serializing on one register group.
__global__ __launch_bounds__(128)
__attribute__((amdgpu_waves_per_eu(4, 8))) void qkv_gemm(
    const bf16* __restrict__ Sb, const bf16* __restrict__ Wb,
    const float* __restrict__ bias, bf16* __restrict__ qb,
    bf16* __restrict__ kb, bf16* __restrict__ vtb) {
  const int lane = threadIdx.x & 31;
  const int wave = threadIdx.x >> 5;
  const int tile = blockIdx.x * 4 + wave;
  const int NtB = N_QKV / 64;  // 48
  const int m0 = (tile / NtB) * 32;
  const int n0 = (tile % NtB) * 64;
  const int ln = lane & 15;
  const int hi = lane >> 4;

  v8f acc[2][4];
#pragma unroll
  for (int j = 0; j < 4; ++j) {
    const float bv = bias[n0 + 16 * j + ln];
#pragma unroll
    for (int i = 0; i < 2; ++i)
#pragma unroll
      for (int r = 0; r < 8; ++r) acc[i][j][r] = bv;
  }

  const bf16* arow0 = Sb + (size_t)(m0 + ln) * DMODEL + 8 * hi;
  const bf16* arow1 = arow0 + (size_t)16 * DMODEL;
  const bf16* brow0 = Wb + (size_t)(n0 + ln) * DMODEL + 8 * hi;

#pragma unroll 8
  for (int kb0 = 0; kb0 < DMODEL; kb0 += 32) {
    // all 6 fragments of this K-step (12 x b128 loads)
    v16bf a0 = pack2(*(const v8bf*)(arow0 + kb0), *(const v8bf*)(arow0 + kb0 + 16));
    v16bf a1 = pack2(*(const v8bf*)(arow1 + kb0), *(const v8bf*)(arow1 + kb0 + 16));
    v16bf bfr[4];
#pragma unroll
    for (int j = 0; j < 4; ++j) {
      const bf16* br = brow0 + (size_t)16 * j * DMODEL + kb0;
      bfr[j] = pack2(*(const v8bf*)br, *(const v8bf*)(br + 16));
    }
    // 8 WMMAs
#pragma unroll
    for (int j = 0; j < 4; ++j) {
      acc[0][j] = wmma_bf16(a0, bfr[j], acc[0][j]);
      acc[1][j] = wmma_bf16(a1, bfr[j], acc[1][j]);
    }
  }

  const int sect = n0 >> 10;        // 0=q,1=k,2=v (uniform per wave)
  const int h = (n0 & 1023) >> 6;   // uniform per wave
#pragma unroll
  for (int i = 0; i < 2; ++i) {
#pragma unroll
    for (int r = 0; r < 8; ++r) {
      const int m = m0 + 16 * i + r + 8 * hi;
      const int b = m >> 11;
      const int t = m & (T_SEQ - 1);
      const size_t bh = (size_t)(b * NH + h);
      if (sect == 0) {
        bf16* row = qb + (bh * T_SEQ + t) * HD;
#pragma unroll
        for (int j = 0; j < 4; ++j) row[16 * j + ln] = (bf16)acc[i][j][r];
      } else if (sect == 1) {
        bf16* row = kb + (bh * T_SEQ + t) * HD;
#pragma unroll
        for (int j = 0; j < 4; ++j) row[16 * j + ln] = (bf16)acc[i][j][r];
      } else {
#pragma unroll
        for (int j = 0; j < 4; ++j)
          vtb[(bh * HD + 16 * j + ln) * T_SEQ + t] = (bf16)acc[i][j][r];
      }
    }
  }
}

// ---------------------------------------------------------------- attention
// One wave per (b,h,16-query tile). Flash online softmax over 32-key chunks
// restricted to the sliding-window band.
__global__ __launch_bounds__(128) void attn_band(
    const bf16* __restrict__ qb, const bf16* __restrict__ kb,
    const bf16* __restrict__ vtb, const int* __restrict__ winp,
    bf16* __restrict__ ctxb) {
  __shared__ __align__(16) bf16 pbuf[4][16 * PLDS_STRIDE];
  const int lane = threadIdx.x & 31;
  const int wave = threadIdx.x >> 5;
  const int tile = blockIdx.x * 4 + wave;       // 0..4095
  const int qt = tile & (T_SEQ / 16 - 1);       // 0..127
  const int bh = tile >> 7;                     // 0..31
  const int q0 = qt * 16;
  const int w = winp[0];
  const int ln = lane & 15;
  const int hi = lane >> 4;
  bf16* plds = pbuf[wave];

  // Q A-fragments (16x64 = two K=32 fragments), resident for the whole loop.
  const bf16* qrow = qb + ((size_t)bh * T_SEQ + q0 + ln) * HD + 8 * hi;
  v16bf aq[2];
#pragma unroll
  for (int j = 0; j < 2; ++j)
    aq[j] = pack2(*(const v8bf*)(qrow + 32 * j),
                  *(const v8bf*)(qrow + 32 * j + 16));

  float mrow[8], lrow[8];
  v8f acc[4];
#pragma unroll
  for (int r = 0; r < 8; ++r) { mrow[r] = -3.0e38f; lrow[r] = 0.0f; }
#pragma unroll
  for (int j = 0; j < 4; ++j) acc[j] = vzero8();

  int kstart = q0 - w; if (kstart < 0) kstart = 0; kstart &= ~31;
  int kend = q0 + 16 + w; if (kend > T_SEQ) kend = T_SEQ;

  const bf16* kbase = kb + (size_t)bh * T_SEQ * HD;
  const bf16* vbase = vtb + (size_t)bh * HD * T_SEQ;
  const float sc = 0.125f;  // 1/sqrt(64)

  for (int kk0 = kstart; kk0 < kend; kk0 += 32) {
    // scores S = Q @ K^T for 32 keys (two 16-col tiles), fp32 accum
    v8f s0 = vzero8(), s1 = vzero8();
    v16bf bk[2][2];
#pragma unroll
    for (int j = 0; j < 2; ++j) {
      const bf16* kr0 = kbase + (size_t)(kk0 + ln) * HD + 32 * j + 8 * hi;
      const bf16* kr1 = kbase + (size_t)(kk0 + 16 + ln) * HD + 32 * j + 8 * hi;
      bk[j][0] = pack2(*(const v8bf*)kr0, *(const v8bf*)(kr0 + 16));
      bk[j][1] = pack2(*(const v8bf*)kr1, *(const v8bf*)(kr1 + 16));
    }
#pragma unroll
    for (int j = 0; j < 2; ++j) {
      s0 = wmma_bf16(aq[j], bk[j][0], s0);
      s1 = wmma_bf16(aq[j], bk[j][1], s1);
    }
    // mask + online softmax per C-row (row = r + 8*hi, cols = ln / ln+16)
#pragma unroll
    for (int r = 0; r < 8; ++r) {
      const int qi = q0 + r + 8 * hi;
      int d0 = qi - (kk0 + ln);      if (d0 < 0) d0 = -d0;
      int d1 = qi - (kk0 + 16 + ln); if (d1 < 0) d1 = -d1;
      const bool ok0 = d0 <= w, ok1 = d1 <= w;
      const float v0 = ok0 ? s0[r] * sc : -3.0e38f;
      const float v1 = ok1 ? s1[r] * sc : -3.0e38f;
      float mx = fmaxf(v0, v1);
#pragma unroll
      for (int off = 8; off > 0; off >>= 1)
        mx = fmaxf(mx, __shfl_xor(mx, off, 32));
      const float mnew = fmaxf(mrow[r], mx);
      const float resc = __expf(mrow[r] - mnew);
      const float p0 = ok0 ? __expf(v0 - mnew) : 0.0f;
      const float p1 = ok1 ? __expf(v1 - mnew) : 0.0f;
      float ps = p0 + p1;
#pragma unroll
      for (int off = 8; off > 0; off >>= 1) ps += __shfl_xor(ps, off, 32);
      lrow[r] = lrow[r] * resc + ps;
      mrow[r] = mnew;
#pragma unroll
      for (int j = 0; j < 4; ++j) acc[j][r] *= resc;
      plds[(r + 8 * hi) * PLDS_STRIDE + ln] = (bf16)p0;
      plds[(r + 8 * hi) * PLDS_STRIDE + ln + 16] = (bf16)p1;
    }
    // LDS is in-order per-wave; barrier stops compiler reordering + drains DS.
    asm volatile("s_wait_dscnt 0" ::: "memory");
    // P (16x32) as A-fragment from LDS
    const bf16* prow = plds + ln * PLDS_STRIDE + 8 * hi;
    v16bf ap = pack2(*(const v8bf*)prow, *(const v8bf*)(prow + 16));
    // acc += P @ V  (V transposed layout -> contiguous B-fragments)
    v16bf bv[4];
#pragma unroll
    for (int j = 0; j < 4; ++j) {
      const bf16* vr = vbase + (size_t)(16 * j + ln) * T_SEQ + kk0 + 8 * hi;
      bv[j] = pack2(*(const v8bf*)vr, *(const v8bf*)(vr + 16));
    }
#pragma unroll
    for (int j = 0; j < 4; ++j) acc[j] = wmma_bf16(ap, bv[j], acc[j]);
  }

  // normalize and scatter ctx as bf16 in (b,t,d) layout
  const int b = bh >> 4;
  const int h = bh & 15;
#pragma unroll
  for (int r = 0; r < 8; ++r) {
    const float inv = 1.0f / lrow[r];
    const int t = q0 + r + 8 * hi;
    const size_t rowoff = ((size_t)b * T_SEQ + t) * DMODEL + h * HD;
#pragma unroll
    for (int j = 0; j < 4; ++j)
      ctxb[rowoff + 16 * j + ln] = (bf16)(acc[j][r] * inv);
  }
}

// ---------------------------------------------------------------- out GEMM
// out[m,n] = sum_k ctx[m,k] * Wout[n,k] + b[n]  (fp32 output).
// Same 2x4 register blocking as qkv_gemm.
__global__ __launch_bounds__(128)
__attribute__((amdgpu_waves_per_eu(4, 8))) void out_gemm(
    const bf16* __restrict__ Cb, const bf16* __restrict__ Wb,
    const float* __restrict__ bias, float* __restrict__ out) {
  const int lane = threadIdx.x & 31;
  const int wave = threadIdx.x >> 5;
  const int tile = blockIdx.x * 4 + wave;
  const int NtB = DMODEL / 64;  // 16
  const int m0 = (tile / NtB) * 32;
  const int n0 = (tile % NtB) * 64;
  const int ln = lane & 15;
  const int hi = lane >> 4;

  v8f acc[2][4];
#pragma unroll
  for (int j = 0; j < 4; ++j) {
    const float bv = bias[n0 + 16 * j + ln];
#pragma unroll
    for (int i = 0; i < 2; ++i)
#pragma unroll
      for (int r = 0; r < 8; ++r) acc[i][j][r] = bv;
  }

  const bf16* arow0 = Cb + (size_t)(m0 + ln) * DMODEL + 8 * hi;
  const bf16* arow1 = arow0 + (size_t)16 * DMODEL;
  const bf16* brow0 = Wb + (size_t)(n0 + ln) * DMODEL + 8 * hi;

#pragma unroll 8
  for (int kb0 = 0; kb0 < DMODEL; kb0 += 32) {
    v16bf a0 = pack2(*(const v8bf*)(arow0 + kb0), *(const v8bf*)(arow0 + kb0 + 16));
    v16bf a1 = pack2(*(const v8bf*)(arow1 + kb0), *(const v8bf*)(arow1 + kb0 + 16));
    v16bf bfr[4];
#pragma unroll
    for (int j = 0; j < 4; ++j) {
      const bf16* br = brow0 + (size_t)16 * j * DMODEL + kb0;
      bfr[j] = pack2(*(const v8bf*)br, *(const v8bf*)(br + 16));
    }
#pragma unroll
    for (int j = 0; j < 4; ++j) {
      acc[0][j] = wmma_bf16(a0, bfr[j], acc[0][j]);
      acc[1][j] = wmma_bf16(a1, bfr[j], acc[1][j]);
    }
  }

#pragma unroll
  for (int i = 0; i < 2; ++i)
#pragma unroll
    for (int r = 0; r < 8; ++r) {
      const int m = m0 + 16 * i + r + 8 * hi;
      float* row = out + (size_t)m * DMODEL + n0;
#pragma unroll
      for (int j = 0; j < 4; ++j) row[16 * j + ln] = acc[i][j][r];
    }
}

// ---------------------------------------------------------------- launch
extern "C" void kernel_launch(void* const* d_in, const int* in_sizes, int n_in,
                              void* d_out, int out_size, void* d_ws,
                              size_t ws_size, hipStream_t stream) {
  const float* S    = (const float*)d_in[0];   // (B,T,d)
  const float* Wqkv = (const float*)d_in[1];   // (3d,d)
  const float* bqkv = (const float*)d_in[2];   // (3d,)
  const float* Wout = (const float*)d_in[3];   // (d,d)
  const float* bout = (const float*)d_in[4];   // (d,)
  const int*   win  = (const int*)d_in[5];     // scalar window
  float* out = (float*)d_out;

  char* ws = (char*)d_ws;
  const size_t sz_S    = (size_t)MROWS * DMODEL * sizeof(bf16);   // 8 MB
  const size_t sz_Wq   = (size_t)N_QKV * DMODEL * sizeof(bf16);   // 6 MB
  const size_t sz_Wo   = (size_t)DMODEL * DMODEL * sizeof(bf16);  // 2 MB
  const size_t sz_head = (size_t)BATCH * NH * T_SEQ * HD * sizeof(bf16);  // 8 MB
  bf16* Sb   = (bf16*)(ws);
  bf16* Wqb  = (bf16*)(ws + sz_S);
  bf16* Wob  = (bf16*)(ws + sz_S + sz_Wq);
  bf16* qb   = (bf16*)(ws + sz_S + sz_Wq + sz_Wo);
  bf16* kb   = (bf16*)(ws + sz_S + sz_Wq + sz_Wo + sz_head);
  bf16* vtb  = (bf16*)(ws + sz_S + sz_Wq + sz_Wo + 2 * sz_head);
  bf16* ctxb = (bf16*)(ws + sz_S + sz_Wq + sz_Wo + 3 * sz_head);

  cvt_f32_bf16<<<4096, 256, 0, stream>>>(S, Sb, MROWS * DMODEL);
  cvt_f32_bf16<<<4096, 256, 0, stream>>>(Wqkv, Wqb, N_QKV * DMODEL);
  cvt_f32_bf16<<<2048, 256, 0, stream>>>(Wout, Wob, DMODEL * DMODEL);

  // (4096/32)*(3072/64) = 6144 waves, 4 waves/block
  qkv_gemm<<<1536, 128, 0, stream>>>(Sb, Wqb, bqkv, qb, kb, vtb);
  // B*H*(T/16) = 4096 query tiles, 4 waves/block
  attn_band<<<1024, 128, 0, stream>>>(qb, kb, vtb, win, ctxb);
  // (4096/32)*(1024/64) = 2048 waves, 4 waves/block
  out_gemm<<<512, 128, 0, stream>>>(ctxb, Wob, bout, out);
}
